// GATModel_54511724920970
// MI455X (gfx1250) — matrix operand order repaired and checked by reading, not compiled, not run on hardware
//
#include <hip/hip_runtime.h>
#include <math.h>

// ---------------------------------------------------------------------------
// GAT (2-layer, PyG GATConv semantics) for gfx1250.
//  - GEMM1: v_wmma_f32_16x16x32_bf16 with A-panel staged in LDS via
//           global_load_async_to_lds_b128 (+ s_wait_asynccnt), ds_load reads
//  - GEMM2: generic wave-per-16x16-tile WMMA
//  - edge softmax via atomic segment max/sum, wave-per-edge aggregation
// ---------------------------------------------------------------------------

#define NEG_SLOPE 0.2f

typedef __attribute__((ext_vector_type(16))) __bf16 v16bf;
typedef __attribute__((ext_vector_type(8)))  float  v8f;

__device__ __forceinline__ unsigned short f32_to_bf16(float f) {
  unsigned int u = __float_as_uint(f);
  u += 0x7FFFu + ((u >> 16) & 1u);          // round-to-nearest-even
  return (unsigned short)(u >> 16);
}

// float -> bf16 elementwise
__global__ void k_cvt_bf16(const float* __restrict__ x,
                           unsigned short* __restrict__ xb, size_t n) {
  size_t i = (size_t)blockIdx.x * blockDim.x + threadIdx.x;
  if (i < n) xb[i] = f32_to_bf16(x[i]);
}

// W[K][Nsrc] (row-major fp32) -> Wt[Npad][K] (bf16, column-major of W, zero pad)
__global__ void k_cvt_transpose(const float* __restrict__ W,
                                unsigned short* __restrict__ Wt,
                                int K, int Nsrc, int Npad) {
  int i = blockIdx.x * blockDim.x + threadIdx.x;
  if (i >= Npad * K) return;
  int n = i / K, k = i % K;
  Wt[i] = (n < Nsrc) ? f32_to_bf16(W[(size_t)k * Nsrc + n]) : (unsigned short)0;
}

__global__ void k_fill_f32(float* __restrict__ p, float v, size_t n) {
  size_t i = (size_t)blockIdx.x * blockDim.x + threadIdx.x;
  if (i < n) p[i] = v;
}

// ---------------------------------------------------------------------------
// GEMM1 (Ncols == 256): block = 8 waves = one 16-row stripe x all 256 cols.
// A panel (16 rows x 32 K, 1KB) staged in LDS with async-to-LDS DMA once per
// K-step; each wave computes two 16x16 tiles (one A frag -> two WMMAs).
// ---------------------------------------------------------------------------
__global__ void k_gemm1_wmma_lds(const unsigned short* __restrict__ A,
                                 const unsigned short* __restrict__ Bt,
                                 float* __restrict__ C,
                                 int M, int K) {
  __shared__ unsigned int sA[16 * 16];        // 16 rows x 32 bf16 = 1KB
  const int tm     = blockIdx.x;
  const int waveId = threadIdx.x >> 5;
  const int lane   = threadIdx.x & 31;
  const int half   = lane >> 4;
  const int l15    = lane & 15;
  const int col0   = (waveId * 2) * 16 + l15;
  const int col1   = col0 + 16;

  // async staging assignment: threads 0..63 copy 16B each (row, 16B chunk)
  const int t      = threadIdx.x;
  const int arow   = (t >> 2) & 15;
  const int achunk = t & 3;
  int growr = tm * 16 + arow; if (growr >= M) growr = M - 1;
  const unsigned short* Ag = A + (size_t)growr * K + achunk * 8;
  // flat->LDS: low 32 bits of the flat shared address are the DS byte offset
  const unsigned int aLds = (unsigned int)(size_t)(&sA[arow * 16 + achunk * 4]);

  const unsigned short* B0 = Bt + (size_t)col0 * K + (half ? 16 : 0);
  const unsigned short* B1 = Bt + (size_t)col1 * K + (half ? 16 : 0);

  v8f acc0 = {0.f, 0.f, 0.f, 0.f, 0.f, 0.f, 0.f, 0.f};
  v8f acc1 = {0.f, 0.f, 0.f, 0.f, 0.f, 0.f, 0.f, 0.f};

  for (int kb = 0; kb < K; kb += 32) {
    if (t < 64) {
      unsigned long long ga = (unsigned long long)(const void*)(Ag + kb);
      asm volatile("global_load_async_to_lds_b128 %0, %1, off"
                   :: "v"(aLds), "v"(ga) : "memory");
      asm volatile("s_wait_asynccnt 0x0" ::: "memory");
    }
    __syncthreads();

    union { v16bf v; unsigned int u[8]; } a, b0, b1;
    // A fragment from LDS (contiguous k order, matching B):
    //  lanes 0-15 : K=0..15 ; lanes 16-31: K=16..31
    const unsigned int* ar = &sA[l15 * 16 + half * 8];
    #pragma unroll
    for (int i = 0; i < 4; ++i) a.u[i]     = ar[i];
    #pragma unroll
    for (int i = 0; i < 4; ++i) a.u[4 + i] = ar[4 + i];
    // B fragments from global (Bt is column-major => pairs contiguous)
    #pragma unroll
    for (int i = 0; i < 8; ++i) b0.u[i] = *(const unsigned int*)(B0 + kb + 2 * i);
    #pragma unroll
    for (int i = 0; i < 8; ++i) b1.u[i] = *(const unsigned int*)(B1 + kb + 2 * i);
    if (kb + 32 < K) {
      __builtin_prefetch(B0 + kb + 32, 0, 1);   // global_prefetch_b8
      __builtin_prefetch(B1 + kb + 32, 0, 1);
    }

    acc0 = __builtin_amdgcn_wmma_f32_16x16x32_bf16(
        false, a.v, false, b0.v, (short)0, acc0, false, false);
    acc1 = __builtin_amdgcn_wmma_f32_16x16x32_bf16(
        false, a.v, false, b1.v, (short)0, acc1, false, false);

    __syncthreads();   // protect sA before next overwrite
  }

  // C/D layout: VGPR r -> M = r + 8*half, N = lane%16
  const int mbase = tm * 16 + (half ? 8 : 0);
  #pragma unroll
  for (int r = 0; r < 8; ++r) {
    int m = mbase + r;
    if (m < M) {
      C[(size_t)m * 256 + col0] = acc0[r];
      C[(size_t)m * 256 + col1] = acc1[r];
    }
  }
}

// ---------------------------------------------------------------------------
// Generic bf16 WMMA GEMM (used for GEMM2): one wave per 16x16 tile.
// C[M][ldC] (f32) = A[M][K] (bf16 row-major) * Bt^T, Bt is [Ncols][K] bf16.
// ---------------------------------------------------------------------------
__global__ void k_gemm_bf16_wmma(const unsigned short* __restrict__ A,
                                 const unsigned short* __restrict__ Bt,
                                 float* __restrict__ C,
                                 int M, int K, int Ncols, int ldC) {
  int wave = (int)((blockIdx.x * (size_t)blockDim.x + threadIdx.x) >> 5);
  int lane = threadIdx.x & 31;
  int tilesN = Ncols >> 4;
  int tm = wave / tilesN;
  int tn = wave - tm * tilesN;
  if (tm * 16 >= M) return;

  int half = lane >> 4;
  int l15  = lane & 15;
  int row  = tm * 16 + l15;
  if (row >= M) row = M - 1;
  int col  = tn * 16 + l15;

  const unsigned short* Ar = A  + (size_t)row * K;
  const unsigned short* Bc = Bt + (size_t)col * K + (half ? 16 : 0);

  v8f acc = {0.f, 0.f, 0.f, 0.f, 0.f, 0.f, 0.f, 0.f};

  for (int kb = 0; kb < K; kb += 32) {
    union { v16bf v; unsigned int u[8]; } a, b;
    #pragma unroll
    for (int i = 0; i < 4; ++i)
      a.u[i]     = *(const unsigned int*)(Ar + kb + 2 * i + (half ? 16 : 0));
    #pragma unroll
    for (int i = 0; i < 4; ++i)
      a.u[4 + i] = *(const unsigned int*)(Ar + kb + 8 + 2 * i + (half ? 16 : 0));
    #pragma unroll
    for (int i = 0; i < 8; ++i)
      b.u[i] = *(const unsigned int*)(Bc + kb + 2 * i);

    if (kb + 32 < K) {
      __builtin_prefetch(Ar + kb + 32, 0, 1);
      __builtin_prefetch(Bc + kb + 32, 0, 1);
    }

    acc = __builtin_amdgcn_wmma_f32_16x16x32_bf16(
        false, a.v, false, b.v, (short)0, acc, false, false);
  }

  int mbase = tm * 16 + (half ? 8 : 0);
  #pragma unroll
  for (int r = 0; r < 8; ++r) {
    int m = mbase + r;
    if (m < M) C[(size_t)m * ldC + col] = acc[r];
  }
}

// per-node attention scores: als[n,h] = <h[n,h,:], a_s[h,:]>, ald likewise
template <int H, int D, int LD>
__global__ void k_scores(const float* __restrict__ h,
                         const float* __restrict__ a_s,
                         const float* __restrict__ a_d,
                         float* __restrict__ als, float* __restrict__ ald,
                         int Nn) {
  int t = blockIdx.x * blockDim.x + threadIdx.x;
  if (t >= Nn * H) return;
  int n = t / H, hh = t - n * H;
  const float* row = h + (size_t)n * LD + hh * D;
  float ss = 0.f, sd = 0.f;
  #pragma unroll 8
  for (int d = 0; d < D; ++d) {
    float v = row[d];
    ss += v * a_s[hh * D + d];
    sd += v * a_d[hh * D + d];
  }
  als[t] = ss;
  ald[t] = sd;
}

__device__ __forceinline__ void atomicMaxF(float* addr, float val) {
  if (val >= 0.f) atomicMax((int*)addr, __float_as_int(val));
  else            atomicMin((unsigned int*)addr, __float_as_uint(val));
}

__device__ __forceinline__ float leaky(float v) {
  return v > 0.f ? v : NEG_SLOPE * v;
}

// pass A: segment max of leaky_relu(als[src]+ald[dst]) by dst
__global__ void k_edge_max(const int* __restrict__ ei, int E, int H,
                           const float* __restrict__ als,
                           const float* __restrict__ ald,
                           float* __restrict__ mx) {
  int t = blockIdx.x * blockDim.x + threadIdx.x;
  if (t >= E * H) return;
  int e = t / H, h = t - e * H;
  int s = ei[e], d = ei[E + e];
  atomicMaxF(&mx[d * H + h], leaky(als[s * H + h] + ald[d * H + h]));
}

// pass B: segment sum of exp(e - max) by dst
__global__ void k_edge_sum(const int* __restrict__ ei, int E, int H,
                           const float* __restrict__ als,
                           const float* __restrict__ ald,
                           const float* __restrict__ mx,
                           float* __restrict__ den) {
  int t = blockIdx.x * blockDim.x + threadIdx.x;
  if (t >= E * H) return;
  int e = t / H, h = t - e * H;
  int s = ei[e], d = ei[E + e];
  float v = leaky(als[s * H + h] + ald[d * H + h]);
  atomicAdd(&den[d * H + h], __expf(v - mx[d * H + h]));
}

// pass C: wave-per-edge weighted scatter-add  agg[dst,:] += alpha * h[src,:]
template <int H, int D, int LDH, int LDAGG>
__global__ void k_edge_agg(const int* __restrict__ ei, int E,
                           const float* __restrict__ als,
                           const float* __restrict__ ald,
                           const float* __restrict__ mx,
                           const float* __restrict__ den,
                           const float* __restrict__ hsrc,
                           float* __restrict__ agg) {
  size_t gid = (size_t)blockIdx.x * blockDim.x + threadIdx.x;
  int wave = (int)(gid >> 5);
  int lane = (int)(gid & 31);
  if (wave >= E) return;
  int s = ei[wave], d = ei[E + wave];
  float alpha[H];
  #pragma unroll
  for (int h = 0; h < H; ++h) {
    float v = leaky(als[s * H + h] + ald[d * H + h]);
    alpha[h] = __expf(v - mx[d * H + h]) / (den[d * H + h] + 1e-16f);
  }
  #pragma unroll
  for (int c = lane; c < H * D; c += 32) {
    atomicAdd(&agg[(size_t)d * LDAGG + c], hsrc[(size_t)s * LDH + c] * alpha[c / D]);
  }
}

// out_bf16 = bf16(elu(agg + bias)), row-major [*, Cc]
__global__ void k_bias_elu_bf16(const float* __restrict__ agg,
                                const float* __restrict__ b,
                                unsigned short* __restrict__ outb,
                                size_t total, int Cc) {
  size_t i = (size_t)blockIdx.x * blockDim.x + threadIdx.x;
  if (i >= total) return;
  float v = agg[i] + b[i % Cc];
  v = v > 0.f ? v : (__expf(v) - 1.f);
  outb[i] = f32_to_bf16(v);
}

// out[n,:] = log_softmax(agg[n,:] + b)
__global__ void k_logsoftmax(const float* __restrict__ agg,
                             const float* __restrict__ b,
                             float* __restrict__ out, int Nn, int Cc) {
  int n = blockIdx.x * blockDim.x + threadIdx.x;
  if (n >= Nn) return;
  const float* row = agg + (size_t)n * Cc;
  float mx = -3.4e38f;
  for (int c = 0; c < Cc; ++c) mx = fmaxf(mx, row[c] + b[c]);
  float s = 0.f;
  for (int c = 0; c < Cc; ++c) s += __expf(row[c] + b[c] - mx);
  float l = logf(s);
  float* orow = out + (size_t)n * Cc;
  for (int c = 0; c < Cc; ++c) orow[c] = row[c] + b[c] - mx - l;
}

// ---------------------------------------------------------------------------

extern "C" void kernel_launch(void* const* d_in, const int* in_sizes, int n_in,
                              void* d_out, int out_size, void* d_ws, size_t ws_size,
                              hipStream_t stream) {
  const float* x      = (const float*)d_in[0];
  const int*   ei     = (const int*)d_in[1];     // int32 under default JAX config
  const float* W1     = (const float*)d_in[2];
  const float* a_src1 = (const float*)d_in[3];
  const float* a_dst1 = (const float*)d_in[4];
  const float* b1     = (const float*)d_in[5];
  const float* W2     = (const float*)d_in[6];
  const float* a_src2 = (const float*)d_in[7];
  const float* a_dst2 = (const float*)d_in[8];
  const float* b2     = (const float*)d_in[9];
  float* out = (float*)d_out;

  const int Nn = in_sizes[0] / 512;   // 50000
  const int E  = in_sizes[1] / 2;     // 850000
  const int F1 = 512, H1 = 256;       // layer1: 512 -> 4*64
  const int CC = 40, CP = 48;         // layer2: 256 -> 40 (padded to 48)

  // workspace carve-out
  char* w = (char*)d_ws;
  auto carve = [&](size_t bytes) {
    char* p = w;
    w += (bytes + 255) & ~(size_t)255;
    return (void*)p;
  };
  unsigned short* xb   = (unsigned short*)carve((size_t)Nn * F1 * 2); // reused as h1b
  unsigned short* W1t  = (unsigned short*)carve((size_t)H1 * F1 * 2);
  unsigned short* W2t  = (unsigned short*)carve((size_t)CP * H1 * 2);
  float* h1   = (float*)carve((size_t)Nn * H1 * 4);
  float* agg1 = (float*)carve((size_t)Nn * H1 * 4);
  float* h2   = (float*)carve((size_t)Nn * CP * 4);
  float* agg2 = (float*)carve((size_t)Nn * CC * 4);
  float* als1 = (float*)carve((size_t)Nn * 4 * 4);
  float* ald1 = (float*)carve((size_t)Nn * 4 * 4);
  float* m1   = (float*)carve((size_t)Nn * 4 * 4);
  float* den1 = (float*)carve((size_t)Nn * 4 * 4);
  float* als2 = (float*)carve((size_t)Nn * 4);
  float* ald2 = (float*)carve((size_t)Nn * 4);
  float* m2   = (float*)carve((size_t)Nn * 4);
  float* den2 = (float*)carve((size_t)Nn * 4);
  unsigned short* h1b = xb;   // xb dead after GEMM1

  const int T = 256;
  auto cdiv = [](long long a, long long b) { return (unsigned)((a + b - 1) / b); };

  // ---- layer 1 ----
  k_cvt_bf16<<<cdiv((long long)Nn * F1, T), T, 0, stream>>>(x, xb, (size_t)Nn * F1);
  k_cvt_transpose<<<cdiv((long long)H1 * F1, T), T, 0, stream>>>(W1, W1t, F1, H1, H1);
  k_cvt_transpose<<<cdiv((long long)CP * H1, T), T, 0, stream>>>(W2, W2t, H1, CC, CP);

  // one block per 16-row stripe; block covers all 256 output cols
  k_gemm1_wmma_lds<<<cdiv(Nn, 16), T, 0, stream>>>(xb, W1t, h1, Nn, F1);

  k_scores<4, 64, 256><<<cdiv((long long)Nn * 4, T), T, 0, stream>>>(
      h1, a_src1, a_dst1, als1, ald1, Nn);

  k_fill_f32<<<cdiv((long long)Nn * 4, T), T, 0, stream>>>(m1, -INFINITY, (size_t)Nn * 4);
  k_fill_f32<<<cdiv((long long)Nn * 4, T), T, 0, stream>>>(den1, 0.f, (size_t)Nn * 4);
  k_fill_f32<<<cdiv((long long)Nn * H1, T), T, 0, stream>>>(agg1, 0.f, (size_t)Nn * H1);

  k_edge_max<<<cdiv((long long)E * 4, T), T, 0, stream>>>(ei, E, 4, als1, ald1, m1);
  k_edge_sum<<<cdiv((long long)E * 4, T), T, 0, stream>>>(ei, E, 4, als1, ald1, m1, den1);
  k_edge_agg<4, 64, 256, 256><<<cdiv((long long)E * 32, T), T, 0, stream>>>(
      ei, E, als1, ald1, m1, den1, h1, agg1);

  k_bias_elu_bf16<<<cdiv((long long)Nn * H1, T), T, 0, stream>>>(
      agg1, b1, h1b, (size_t)Nn * H1, H1);

  // ---- layer 2 ----
  long long tiles2 = (long long)((Nn + 15) / 16) * (CP / 16);
  k_gemm_bf16_wmma<<<cdiv(tiles2 * 32, T), T, 0, stream>>>(h1b, W2t, h2, Nn, H1, CP, CP);

  k_scores<1, 40, 48><<<cdiv((long long)Nn, T), T, 0, stream>>>(
      h2, a_src2, a_dst2, als2, ald2, Nn);

  k_fill_f32<<<cdiv((long long)Nn, T), T, 0, stream>>>(m2, -INFINITY, (size_t)Nn);
  k_fill_f32<<<cdiv((long long)Nn, T), T, 0, stream>>>(den2, 0.f, (size_t)Nn);
  k_fill_f32<<<cdiv((long long)Nn * CC, T), T, 0, stream>>>(agg2, 0.f, (size_t)Nn * CC);

  k_edge_max<<<cdiv((long long)E, T), T, 0, stream>>>(ei, E, 1, als2, ald2, m2);
  k_edge_sum<<<cdiv((long long)E, T), T, 0, stream>>>(ei, E, 1, als2, ald2, m2, den2);
  k_edge_agg<1, 40, 48, 40><<<cdiv((long long)E * 32, T), T, 0, stream>>>(
      ei, E, als2, ald2, m2, den2, h2, agg2);

  k_logsoftmax<<<cdiv((long long)Nn, T), T, 0, stream>>>(agg2, b2, out, Nn, CC);
}
